// LSTM_model_77764677861617
// MI455X (gfx1250) — matrix operand order repaired and checked by reading, not compiled, not run on hardware
//
#include <hip/hip_runtime.h>
#include <hip/hip_bf16.h>

typedef __bf16 bf16;
typedef __attribute__((ext_vector_type(16))) __bf16 v16bf;
typedef __attribute__((ext_vector_type(8)))  __bf16 v8bf;
typedef __attribute__((ext_vector_type(8)))  float  v8f;
typedef __attribute__((ext_vector_type(4)))  unsigned uint4v;
typedef __attribute__((ext_vector_type(8)))  int      int8v;
typedef __attribute__((ext_vector_type(4)))  int      int4v;

union V16U { v16bf v; v8bf h[2]; };

constexpr int B = 32, S = 1024, NI = 512, NH = 1024, NO = 512;
constexpr int NGATE = 4 * NH;        // 4096
constexpr int KT_X  = NI / 32;       // 16 k-tiles from x
constexpr int KT_H  = NH / 32;       // 32 k-tiles from h
constexpr int KT_ALL = KT_X + KT_H;  // 48
constexpr int NWG  = 64;             // persistent grid: 1 WG per 16 hidden cols
constexpr int FRAG = 512;            // elems per packed 32x16 bf16 fragment
constexpr int WSLICE = 4 * KT_ALL * FRAG;   // 98304 bf16 = 192 KB per WG

// ---- fragment helpers (CDNA5 16-bit WMMA layouts) ----
__device__ __forceinline__ float sigf(float x) { return 1.0f / (1.0f + __expf(-x)); }

// A fragment from a bf16 row-major row. p = row + kbase + koff (koff = (lane>=16)*8).
__device__ __forceinline__ v16bf a_from_bf16(const bf16* __restrict__ p) {
  V16U r;
  r.h[0] = *(const v8bf*)p;
  r.h[1] = *(const v8bf*)(p + 16);
  return r.v;
}
// A fragment from an fp32 row (used only in out_proj). p = row + kbase + koff.
__device__ __forceinline__ v16bf a_from_f32(const float* __restrict__ p) {
  V16U r;
#pragma unroll
  for (int i = 0; i < 8; ++i) { r.h[0][i] = (bf16)p[i]; r.h[1][i] = (bf16)p[16 + i]; }
  return r.v;
}
// Pre-packed fragment: lane-contiguous 16 bf16 (32 bytes). Works for global and LDS.
__device__ __forceinline__ v16bf pk_frag(const bf16* p) {
  V16U r;
  r.h[0] = *(const v8bf*)p;
  r.h[1] = *(const v8bf*)(p + 8);
  return r.v;
}

// ---- one-shot packers ----
// Bpack[wg][g][kt(48)][lane(32)][j(16)] ; n = g*NH + wg*16 + (lane&15),
// k = kt*32 + (lane>>4)*8 + (j<8 ? j : j+8) ; source row-major [K, 4096] = [W ; U]
__global__ void pack_WU(const float* __restrict__ W, const float* __restrict__ U,
                        bf16* __restrict__ Bpack) {
  int idx = blockIdx.x * 256 + threadIdx.x;
  constexpr int total = NWG * WSLICE;
  if (idx >= total) return;
  int w = idx;
  int j    = w & 15;  w >>= 4;
  int lane = w & 31;  w >>= 5;
  int kt   = w % KT_ALL; w /= KT_ALL;
  int g    = w & 3;   w >>= 2;
  int wg   = w;
  int nn = g * NH + wg * 16 + (lane & 15);
  int k  = kt * 32 + (lane >> 4) * 8 + (j < 8 ? j : j + 8);
  float v = (k < NI) ? W[(size_t)k * NGATE + nn] : U[(size_t)(k - NI) * NGATE + nn];
  Bpack[idx] = (bf16)v;
}

// Vpack[nt(32)][kt(32)][lane][j] from V[1024,512]
__global__ void pack_V(const float* __restrict__ V, bf16* __restrict__ Vpack) {
  int idx = blockIdx.x * 256 + threadIdx.x;
  constexpr int total = (NO / 16) * (NH / 32) * FRAG;   // 524288
  if (idx >= total) return;
  int w = idx;
  int j    = w & 15;  w >>= 4;
  int lane = w & 31;  w >>= 5;
  int kt   = w % (NH / 32); w /= (NH / 32);
  int nt   = w;
  int nn = nt * 16 + (lane & 15);
  int k  = kt * 32 + (lane >> 4) * 8 + (j < 8 ? j : j + 8);
  Vpack[idx] = (bf16)V[(size_t)k * NO + nn];
}

// Xpack[t][mt(2)][kt(16)][lane(32)][j(16)]: x pre-converted into A-fragment order.
// For A 16-bit layout: lane holds row m = mt*16 + (lane&15);
// element j maps to k = kt*32 + (lane>=16 ? 8 : 0) + (j<8 ? j : j+8).
__global__ void pack_X(const float* __restrict__ x, bf16* __restrict__ Xpack) {
  int idx = blockIdx.x * 256 + threadIdx.x;
  constexpr int total = S * 2 * KT_X * FRAG;            // 16777216
  if (idx >= total) return;
  int w = idx;
  int j    = w & 15;  w >>= 4;
  int lane = w & 31;  w >>= 5;
  int kt   = w & 15;  w >>= 4;
  int mt   = w & 1;   w >>= 1;
  int t    = w;
  int m = mt * 16 + (lane & 15);
  int k = kt * 32 + ((lane >> 4) * 8) + (j < 8 ? j : j + 8);
  Xpack[idx] = (bf16)x[((size_t)m * S + t) * NI + k];
}

__global__ void init_state(bf16* __restrict__ hbuf, unsigned* __restrict__ counter) {
  int idx = blockIdx.x * 256 + threadIdx.x;
  if (idx < 2 * B * NH) hbuf[idx] = (bf16)0.0f;
  if (idx == 0) *counter = 0u;
}

// ---- persistent LSTM recurrence: 64 WGs x 64 threads (2 waves) ----
// Wave owns one 16x16 hidden tile; all 192 KB of its gate weights live in LDS
// for the whole sequence (staged once via the Tensor Data Mover).
__global__ __launch_bounds__(64) void lstm_recurrence(
    const bf16* __restrict__ Xpack, const float* __restrict__ hid_bias,
    const bf16* __restrict__ Bpack, bf16* __restrict__ hbuf,
    float* __restrict__ out_hidden, float* __restrict__ out_context,
    unsigned* __restrict__ counter) {
  __shared__ __align__(16) bf16 Bs[WSLICE];   // 192 KB, layout == Bpack wg-slice

  const int wg   = blockIdx.x;
  const int tid  = threadIdx.x;
  const int wave = tid >> 5;
  const int lane = tid & 31;
  const int mbase  = wave * 16;
  const int ncol   = lane & 15;
  const int koff   = (lane < 16) ? 0 : 8;
  const int rowoff = (lane >> 4) * 8;
  const int n      = wg * 16 + ncol;

  const bf16* wslice = Bpack + (size_t)wg * WSLICE;

#if __has_builtin(__builtin_amdgcn_tensor_load_to_lds)
  // TDM: one 192KB contiguous DMA, global -> LDS.
  if (wave == 0) {
    unsigned long long ga = (unsigned long long)(uintptr_t)wslice;
    // low 32 bits of a generic pointer into LDS == LDS byte offset
    unsigned lds_off = (unsigned)(uintptr_t)&Bs[0];
    constexpr unsigned UNITS = (unsigned)(WSLICE * sizeof(bf16) / 8);  // 24576 x 8B
    uint4v g0;
    g0[0] = 1u;                                               // count=1, user descriptor
    g0[1] = lds_off;                                          // lds_addr (bytes)
    g0[2] = (unsigned)(ga & 0xffffffffu);                     // global_addr[31:0]
    g0[3] = (unsigned)((ga >> 32) & 0x01ffffffu) | (2u << 30);// global_addr[56:32] | type=2
    int8v g1;
    g1[0] = (int)(3u << 16);       // workgroup_mask=0, data_size=3 (8B)
    g1[1] = (int)(UNITS << 16);    // tensor_dim0[15:0]
    g1[2] = (int)(1u << 16);       // tensor_dim0[31:16]=0 | tensor_dim1=1
    g1[3] = (int)(UNITS << 16);    // tensor_dim1[31:16]=0 | tile_dim0
    g1[4] = 1;                     // tile_dim1=1, tile_dim2=0
    g1[5] = (int)UNITS;            // tensor_dim0_stride[31:0]
    g1[6] = 0;                     // stride hi | tensor_dim1_stride lo
    g1[7] = 0;
    int4v z4 = {0, 0, 0, 0};
    int8v z8 = {0, 0, 0, 0, 0, 0, 0, 0};
    __builtin_amdgcn_tensor_load_to_lds(g0, g1, z4, z4, z8, 0);
    __builtin_amdgcn_s_wait_tensorcnt(0);
  }
#else
  for (int e = tid; e < WSLICE / 8; e += 64)
    *(v8bf*)&Bs[e * 8] = *(const v8bf*)&wslice[e * 8];
#endif
  // The TDM builtin writes LDS through a descriptor the compiler can't see.
  // Escape Bs into an asm memory clobber so ds_loads of Bs aren't folded to undef.
  {
    bf16* bsp = &Bs[0];
    asm volatile("" : : "v"(bsp) : "memory");
  }
  __syncthreads();

  float bias[4];
#pragma unroll
  for (int g = 0; g < 4; ++g) bias[g] = hid_bias[g * NH + n];

  float creg[8];
#pragma unroll
  for (int r = 0; r < 8; ++r) creg[r] = 0.0f;

  for (int t = 0; t < S; ++t) {
    const bf16* hR = hbuf + (size_t)(t & 1) * B * NH;
    bf16*       hW = hbuf + (size_t)((t + 1) & 1) * B * NH;

    v8f acc[4];
#pragma unroll
    for (int g = 0; g < 4; ++g) {
      v8f a;
#pragma unroll
      for (int r = 0; r < 8; ++r) a[r] = bias[g];
      acc[g] = a;
    }

    // x contribution: pre-packed bf16 A fragments, weights from LDS
    const bf16* xfr = Xpack + (((size_t)(t * 2 + wave)) * KT_X) * FRAG + lane * 16;
#pragma unroll 4
    for (int kt = 0; kt < KT_X; ++kt) {
      v16bf a = pk_frag(xfr + (size_t)kt * FRAG);
#pragma unroll
      for (int g = 0; g < 4; ++g) {
        v16bf b = pk_frag(&Bs[(g * KT_ALL + kt) * FRAG + lane * 16]);
        acc[g] = __builtin_amdgcn_wmma_f32_16x16x32_bf16(false, a, false, b,
                                                         (short)0, acc[g], false, false);
      }
    }

    // h contribution: row-major bf16 h, weights from LDS
    const bf16* hRow = hR + (size_t)(mbase + ncol) * NH;
#pragma unroll 4
    for (int kt = 0; kt < KT_H; ++kt) {
      v16bf a = a_from_bf16(hRow + kt * 32 + koff);
#pragma unroll
      for (int g = 0; g < 4; ++g) {
        v16bf b = pk_frag(&Bs[(g * KT_ALL + KT_X + kt) * FRAG + lane * 16]);
        acc[g] = __builtin_amdgcn_wmma_f32_16x16x32_bf16(false, a, false, b,
                                                         (short)0, acc[g], false, false);
      }
    }

    // LSTM cell (fp32) + stores. C-tile layout: row = r + (lane>=16 ? 8 : 0), col = lane&15.
#pragma unroll
    for (int r = 0; r < 8; ++r) {
      int bb = mbase + r + rowoff;
      float iv = sigf(acc[0][r]);
      float fv = sigf(acc[1][r]);
      float gv = tanhf(acc[2][r]);
      float ov = sigf(acc[3][r]);
      float cn = fv * creg[r] + iv * gv;
      float hn = ov * tanhf(cn);
      creg[r] = cn;
      size_t o = ((size_t)bb * S + t) * NH + n;
      out_hidden[o]  = hn;
      out_context[o] = cn;
      hW[bb * NH + n] = (bf16)hn;
    }

    // device-wide step barrier (monotonic counter, zeroed by init_state each call)
    __threadfence();
    __syncthreads();
    if (tid == 0) {
      atomicAdd(counter, 1u);
      unsigned target = (unsigned)(t + 1) * (unsigned)NWG;
      while (__hip_atomic_load(counter, __ATOMIC_RELAXED, __HIP_MEMORY_SCOPE_AGENT) < target) {
        __builtin_amdgcn_s_sleep(2);
      }
    }
    __syncthreads();
    __threadfence();
  }
}

// ---- output projection: [32768,1024] x [1024,512] bf16 WMMA, one 16x16 tile per wave ----
__global__ __launch_bounds__(256) void out_proj(
    const float* __restrict__ hidden, const bf16* __restrict__ Vpack,
    const float* __restrict__ out_bias, float* __restrict__ out) {
  const int lane = threadIdx.x & 31;
  const int wgl  = blockIdx.x * 8 + (threadIdx.x >> 5);
  const int mt = wgl >> 5;          // 0..2047
  const int nt = wgl & 31;          // 0..31
  const int ncol   = lane & 15;
  const int koff   = (lane < 16) ? 0 : 8;
  const int rowoff = (lane >> 4) * 8;
  const int n = nt * 16 + ncol;

  v8f acc;
  float bv = out_bias[n];
#pragma unroll
  for (int r = 0; r < 8; ++r) acc[r] = bv;

  const float* aRow = hidden + (size_t)(mt * 16 + ncol) * NH;
#pragma unroll 4
  for (int kt = 0; kt < NH / 32; ++kt) {
    v16bf a = a_from_f32(aRow + kt * 32 + koff);
    v16bf b = pk_frag(Vpack + ((size_t)(nt * (NH / 32) + kt)) * FRAG + lane * 16);
    acc = __builtin_amdgcn_wmma_f32_16x16x32_bf16(false, a, false, b,
                                                  (short)0, acc, false, false);
  }
#pragma unroll
  for (int r = 0; r < 8; ++r) {
    size_t ro = (size_t)(mt * 16 + r + rowoff);
    out[ro * NO + n] = acc[r];
  }
}

extern "C" void kernel_launch(void* const* d_in, const int* in_sizes, int n_in,
                              void* d_out, int out_size, void* d_ws, size_t ws_size,
                              hipStream_t stream) {
  (void)in_sizes; (void)n_in; (void)out_size; (void)ws_size;
  const float* x  = (const float*)d_in[0];
  const float* W  = (const float*)d_in[1];
  const float* U  = (const float*)d_in[2];
  const float* hb = (const float*)d_in[3];
  const float* V  = (const float*)d_in[4];
  const float* ob = (const float*)d_in[5];

  float* out_hidden  = (float*)d_out;                       // [32,1024,1024]
  float* out_output  = out_hidden + (size_t)B * S * NH;     // [32,1024,512]
  float* out_context = out_output + (size_t)B * S * NO;     // [32,1024,1024]

  char* ws = (char*)d_ws;
  constexpr size_t BPACK_BYTES = (size_t)NWG * WSLICE * sizeof(bf16);                 // 12 MB
  constexpr size_t VPACK_BYTES = (size_t)(NO / 16) * (NH / 32) * FRAG * sizeof(bf16); // 1 MB
  constexpr size_t HBUF_BYTES  = (size_t)2 * B * NH * sizeof(bf16);                   // 128 KB
  constexpr size_t CNT_BYTES   = 256;
  bf16* Bpack = (bf16*)ws;
  bf16* Vpack = (bf16*)(ws + BPACK_BYTES);
  bf16* hbuf  = (bf16*)(ws + BPACK_BYTES + VPACK_BYTES);
  unsigned* counter = (unsigned*)(ws + BPACK_BYTES + VPACK_BYTES + HBUF_BYTES);
  bf16* Xpack = (bf16*)(ws + BPACK_BYTES + VPACK_BYTES + HBUF_BYTES + CNT_BYTES);     // 32 MB

  {
    constexpr int total = NWG * WSLICE;
    pack_WU<<<(total + 255) / 256, 256, 0, stream>>>(W, U, Bpack);
  }
  {
    constexpr int total = (NO / 16) * (NH / 32) * FRAG;
    pack_V<<<(total + 255) / 256, 256, 0, stream>>>(V, Vpack);
  }
  {
    constexpr int total = S * 2 * KT_X * FRAG;
    pack_X<<<(total + 255) / 256, 256, 0, stream>>>(x, Xpack);
  }
  init_state<<<(2 * B * NH + 255) / 256, 256, 0, stream>>>(hbuf, counter);

  lstm_recurrence<<<NWG, 64, 0, stream>>>(Xpack, hb, Bpack, hbuf,
                                          out_hidden, out_context, counter);

  out_proj<<<(2048 * 32) / 8, 256, 0, stream>>>(out_hidden, Vpack, ob, out_output);
}
